// MultiSequenceEventTokenizer_26792005992816
// MI455X (gfx1250) — matrix operand order repaired and checked by reading, not compiled, not run on hardware
//
#include <hip/hip_runtime.h>
#include <hip/hip_bf16.h>

// Problem constants (from reference)
#define B_      32
#define L_      256
#define S_      4
#define D_      512
#define H_      1024          // L_*S_
#define FIVE_D  2560
#define K1      2560          // GEMM1 K
#define N1      2048          // GEMM1 N (=4*D)
#define K2      2048          // GEMM2 K
#define N2      512           // GEMM2 N (=D)
#define TGB_    128
#define LN_EPS  1e-5f
#define MTILE   32            // event rows per workgroup (2 x 16-row WMMA subtiles)

typedef __attribute__((ext_vector_type(16))) __bf16 v16bf;
typedef __attribute__((ext_vector_type(8)))  float  v8f;

union FragB16 { uint4 u4[2]; v16bf v; };

// ---------------------------------------------------------------------------
// Kernel 1: repack weights f32 -> bf16, transposed to column-K-major so a
// WMMA B-fragment lane load is two contiguous 16B (8-K) chunks, one per 16-K half.
// w1b[n*K1 + k] = bf16(w1[k*N1 + n]);  w2b[n*K2 + k] = bf16(w2[k*N2 + n])
// ---------------------------------------------------------------------------
__global__ void convert_weights(const float* __restrict__ w1,
                                const float* __restrict__ w2,
                                __bf16* __restrict__ w1b,
                                __bf16* __restrict__ w2b) {
  const long stride = (long)gridDim.x * blockDim.x;
  const long tid0 = (long)blockIdx.x * blockDim.x + threadIdx.x;
  const long tot1 = (long)K1 * N1;
  for (long o = tid0; o < tot1; o += stride) {
    int n = (int)(o / K1), k = (int)(o % K1);
    w1b[o] = (__bf16)w1[(long)k * N1 + n];
  }
  const long tot2 = (long)K2 * N2;
  for (long o = tid0; o < tot2; o += stride) {
    int n = (int)(o / K2), k = (int)(o % K2);
    w2b[o] = (__bf16)w2[(long)k * N2 + n];
  }
}

// ---------------------------------------------------------------------------
// Kernel 2: zero the states output (masked-off slots must be exactly 0).
// ---------------------------------------------------------------------------
__global__ void zero_states(float4* __restrict__ p, long n4) {
  const long stride = (long)gridDim.x * blockDim.x;
  for (long i = (long)blockIdx.x * blockDim.x + threadIdx.x; i < n4; i += stride)
    p[i] = make_float4(0.f, 0.f, 0.f, 0.f);
}

// ---------------------------------------------------------------------------
// Kernel 3: per-batch bucketization. One workgroup per batch.
// dest[b*H + h] = flattened (b*S + s)*L + slot, or -1 if dropped.
// Also writes sequence_mask (as 0/1 float) and the empty-group slot-0 base.
// ---------------------------------------------------------------------------
__global__ void bucketize(const int* __restrict__ group_ids,
                          const int* __restrict__ lengths,
                          const float* __restrict__ pos_emb,
                          const float* __restrict__ seq_id_emb,
                          const float* __restrict__ empty_tokens,
                          int*   __restrict__ dest,
                          float* __restrict__ states,
                          float* __restrict__ mask_out) {
  const int b = blockIdx.x;
  __shared__ int cnt[S_];
  if (threadIdx.x == 0) {
    const int len = lengths[b];
    int c[S_] = {0, 0, 0, 0};
    for (int h = 0; h < H_; ++h) {
      if (h < len) {
        int g = group_ids[b * H_ + h];
        if (g >= 1 && g <= S_) c[g - 1]++;
      }
    }
    int off[S_], r[S_];
    for (int s = 0; s < S_; ++s) {
      off[s] = c[s] > L_ ? c[s] - L_ : 0;
      r[s] = 0;
      cnt[s] = c[s];
    }
    for (int h = 0; h < H_; ++h) {
      int d = -1;
      if (h < len) {
        int g = group_ids[b * H_ + h];
        if (g >= 1 && g <= S_) {
          int rk = r[g - 1]++;
          if (rk >= off[g - 1])
            d = (b * S_ + (g - 1)) * L_ + (rk - off[g - 1]);
        }
      }
      dest[b * H_ + h] = d;
    }
  }
  __syncthreads();

  // sequence_mask as float 0/1
  for (int idx = threadIdx.x; idx < S_ * L_; idx += blockDim.x) {
    int s = idx / L_, l = idx % L_;
    int c = cnt[s];
    int kept = (c == 0) ? 1 : (c < L_ ? c : L_);
    mask_out[b * S_ * L_ + idx] = (l < kept) ? 1.0f : 0.0f;
  }
  // empty groups: slot 0 gets empty_token + pos_emb[0] + seq_id_emb[s+1]
  for (int d = threadIdx.x; d < S_ * D_; d += blockDim.x) {
    int s = d / D_, j = d % D_;
    if (cnt[s] == 0) {
      states[((long)(b * S_ + s) * L_) * D_ + j] =
          empty_tokens[s * D_ + j] + pos_emb[j] + seq_id_emb[(s + 1) * D_ + j];
    }
  }
}

// ---------------------------------------------------------------------------
// WMMA fragment loaders (layouts per CDNA5 ISA 7.12.2, 16-bit matrices)
// A (16x32, MxK): lane -> m = lane&15;  K chunks [kb, kb+8) and [kb+16, kb+24),
//                 kb = (lane>>4)*8
// B (32x16, KxN): lane -> n = lane&15;  K chunks [kb, kb+8) and [kb+16, kb+24),
//                 kb = (lane>>4)*8    (weights stored column-K-major)
// ---------------------------------------------------------------------------
__device__ inline v16bf load_a_frag(const __bf16* rowp, int kbase, int kbA) {
  FragB16 f;
  f.u4[0] = *(const uint4*)(rowp + kbase + kbA);
  f.u4[1] = *(const uint4*)(rowp + kbase + kbA + 16);
  return f.v;
}
__device__ inline v16bf load_b_frag(const __bf16* colp, int kbase) {
  FragB16 f;
  f.u4[0] = *(const uint4*)(colp + kbase);
  f.u4[1] = *(const uint4*)(colp + kbase + 16);
  return f.v;
}

__device__ inline float silu_fast(float v) {
  // v * sigmoid(v) with fast hardware rcp (avoids IEEE div_scale sequence)
  return v * __builtin_amdgcn_rcpf(1.0f + __expf(-v));
}

// ---------------------------------------------------------------------------
// Kernel 4: fused gather + LayerNorm + MLP (two bf16 WMMA GEMMs) + scatter.
// One workgroup (8 wave32s) per 32 event rows; the two 16-row WMMA subtiles
// share every B fragment (halves L2 weight traffic vs M=16).
// LDS layouts are msub-interleaved: x[16][2][K1], h[16][2][K2] so both
// subtiles of row m sit within DS 16-bit immediate-offset reach of ONE base
// VGPR -> no per-iteration address VALU in the K loops.
// LDS: 160KB + 128KB = 288KB (< 320KB/WGP).
// ---------------------------------------------------------------------------
__global__ __launch_bounds__(256, 1)
void event_mlp_scatter(const float* __restrict__ embed_table,
                       const float* __restrict__ time_gap_emb,
                       const float* __restrict__ seq_id_emb,
                       const float* __restrict__ pos_emb,
                       const float* __restrict__ ln_gamma,
                       const float* __restrict__ ln_beta,
                       const __bf16* __restrict__ w1b,
                       const float* __restrict__ b1,
                       const __bf16* __restrict__ w2b,
                       const float* __restrict__ b2,
                       const int* __restrict__ history_tokens,
                       const int* __restrict__ post_tokens,
                       const int* __restrict__ author_tokens,
                       const int* __restrict__ action_tokens,
                       const int* __restrict__ time_gap,
                       const int* __restrict__ lengths,
                       const int* __restrict__ dest,
                       float* __restrict__ states) {
  __shared__ __bf16 xlds[16][2][FIVE_D];  // [m][msub][k]  (160 KB)
  __shared__ __bf16 hlds[16][2][N1];      // [m][msub][k]  (128 KB)
  __shared__ int dest_s[MTILE];

  const int tid = threadIdx.x;
  const int lane = tid & 31;
  const int wv = tid >> 5;
  const int rowBase = blockIdx.x * MTILE;

  if (tid < MTILE) dest_s[tid] = dest[rowBase + tid];

  // ------------- Phase 0: embedding gather + LayerNorm -> bf16 LDS ---------
  for (int rr = 0; rr < MTILE / 8; ++rr) {
    const int row = wv * (MTILE / 8) + rr;   // logical row 0..31
    const int e = rowBase + row;
    const int b = e >> 10;       // / H_
    const int h = e & (H_ - 1);
    const int len = lengths[b];
    const bool valid = h < len;
    const int t0 = valid ? history_tokens[e] : 0;
    const int t1 = valid ? post_tokens[e] : 0;
    const int t2 = valid ? author_tokens[e] : 0;
    const int t3 = valid ? action_tokens[e] : 0;
    int tg = valid ? time_gap[e] : 0;
    tg = tg < 0 ? 0 : (tg > TGB_ ? TGB_ : tg);

    const float* p0 = embed_table + (long)t0 * D_;
    const float* p1 = embed_table + (long)t1 * D_;
    const float* p2 = embed_table + (long)t2 * D_;
    const float* p3 = embed_table + (long)t3 * D_;
    const float* p4 = time_gap_emb + (long)tg * D_;

    float sum = 0.f, ssq = 0.f;
    for (int k = lane; k < FIVE_D; k += 32) {
      const int ei = k >> 9, d = k & (D_ - 1);
      const float* p = (ei == 0) ? p0 : (ei == 1) ? p1 : (ei == 2) ? p2
                                      : (ei == 3) ? p3 : p4;
      const float v = p[d];
      sum += v;
      ssq += v * v;
    }
#pragma unroll
    for (int off = 16; off >= 1; off >>= 1) {
      sum += __shfl_xor(sum, off, 32);
      ssq += __shfl_xor(ssq, off, 32);
    }
    const float mu = sum * (1.0f / FIVE_D);
    const float var = ssq * (1.0f / FIVE_D) - mu * mu;
    const float rs = rsqrtf(var + LN_EPS);

    __bf16* xrow = &xlds[row & 15][row >> 4][0];
    for (int k = lane; k < FIVE_D; k += 32) {
      const int ei = k >> 9, d = k & (D_ - 1);
      const float* p = (ei == 0) ? p0 : (ei == 1) ? p1 : (ei == 2) ? p2
                                      : (ei == 3) ? p3 : p4;
      const float nx = (p[d] - mu) * rs * ln_gamma[k] + ln_beta[k];
      xrow[k] = (__bf16)nx;
    }
  }
  __syncthreads();

  const int m     = lane & 15;
  const int kbA   = (lane >> 4) * 8;
  const int kbB   = (lane >> 4) * 16;
  const int ncol  = lane & 15;
  const int mrow0 = (lane >> 4) * 8;

  // ------------- Phase 1: GEMM1 (32 x 2048 = x @ w1), SiLU, -> hlds --------
  // One B fragment feeds TWO wmmas (msub 0 and 1, +FIVE_D*2B DS offset).
  {
    const __bf16* arow = &xlds[m][0][0];  // msub1 = arow + FIVE_D (immediate)
    for (int t = wv; t < N1 / 16; t += 8) {
      const __bf16* colp = w1b + (long)(t * 16 + ncol) * K1;
      v8f acc0 = {}, acc1 = {};
#pragma unroll 4
      for (int kk = 0; kk < K1; kk += 32) {
        v16bf bb = load_b_frag(colp, kk + kbA);
        v16bf a0 = load_a_frag(arow, kk, kbA);
        v16bf a1 = load_a_frag(arow + FIVE_D, kk, kbA);
        acc0 = __builtin_amdgcn_wmma_f32_16x16x32_bf16(
            false, a0, false, bb, (short)0, acc0, false, false);
        acc1 = __builtin_amdgcn_wmma_f32_16x16x32_bf16(
            false, a1, false, bb, (short)0, acc1, false, false);
      }
      const float bias = b1[t * 16 + ncol];
#pragma unroll
      for (int r = 0; r < 8; ++r) {
        hlds[mrow0 + r][0][t * 16 + ncol] = (__bf16)silu_fast(acc0[r] + bias);
        hlds[mrow0 + r][1][t * 16 + ncol] = (__bf16)silu_fast(acc1[r] + bias);
      }
    }
  }
  __syncthreads();

  // ------------- Phase 2: GEMM2 (32 x 512 = h @ w2) + scatter-store --------
  {
    const __bf16* arow = &hlds[m][0][0];  // msub1 = arow + N1 (immediate)
    for (int t = wv; t < N2 / 16; t += 8) {
      const __bf16* colp = w2b + (long)(t * 16 + ncol) * K2;
      v8f acc0 = {}, acc1 = {};
#pragma unroll 4
      for (int kk = 0; kk < K2; kk += 32) {
        v16bf bb = load_b_frag(colp, kk + kbA);
        v16bf a0 = load_a_frag(arow, kk, kbA);
        v16bf a1 = load_a_frag(arow + N1, kk, kbA);
        acc0 = __builtin_amdgcn_wmma_f32_16x16x32_bf16(
            false, a0, false, bb, (short)0, acc0, false, false);
        acc1 = __builtin_amdgcn_wmma_f32_16x16x32_bf16(
            false, a1, false, bb, (short)0, acc1, false, false);
      }
      const int n = t * 16 + ncol;
      const float bias = b2[n];
#pragma unroll
      for (int r = 0; r < 8; ++r) {
#pragma unroll
        for (int msub = 0; msub < 2; ++msub) {
          const int mrow = msub * 16 + mrow0 + r;
          const int dst = dest_s[mrow];
          if (dst >= 0) {
            const int slot = dst & (L_ - 1);
            const int s = (dst >> 8) & (S_ - 1);  // (dst / L_) % S_
            const float acc = (msub == 0) ? acc0[r] : acc1[r];
            const float v = acc + bias + pos_emb[slot * D_ + n] +
                            seq_id_emb[(s + 1) * D_ + n];
            states[(long)dst * D_ + n] = v;
          }
        }
      }
    }
  }
}

// ---------------------------------------------------------------------------
extern "C" void kernel_launch(void* const* d_in, const int* in_sizes, int n_in,
                              void* d_out, int out_size, void* d_ws, size_t ws_size,
                              hipStream_t stream) {
  const float* embed_table  = (const float*)d_in[0];
  const float* time_gap_emb = (const float*)d_in[1];
  const float* seq_id_emb   = (const float*)d_in[2];
  const float* pos_emb      = (const float*)d_in[3];
  const float* ln_gamma     = (const float*)d_in[4];
  const float* ln_beta      = (const float*)d_in[5];
  const float* w1           = (const float*)d_in[6];
  const float* b1           = (const float*)d_in[7];
  const float* w2           = (const float*)d_in[8];
  const float* b2           = (const float*)d_in[9];
  const float* empty_tokens = (const float*)d_in[10];
  const int* history_tokens = (const int*)d_in[11];
  const int* post_tokens    = (const int*)d_in[12];
  const int* author_tokens  = (const int*)d_in[13];
  const int* action_tokens  = (const int*)d_in[14];
  const int* time_gap       = (const int*)d_in[15];
  const int* group_ids      = (const int*)d_in[16];
  const int* lengths        = (const int*)d_in[17];

  float* states   = (float*)d_out;
  float* mask_out = states + (long)B_ * S_ * L_ * D_;

  char* ws = (char*)d_ws;
  __bf16* w1b = (__bf16*)ws;                                   // 10,485,760 B
  __bf16* w2b = (__bf16*)(ws + (size_t)K1 * N1 * 2);           //  2,097,152 B
  int* dest = (int*)(ws + (size_t)K1 * N1 * 2 + (size_t)K2 * N2 * 2);

  convert_weights<<<512, 256, 0, stream>>>(w1, w2, w1b, w2b);
  zero_states<<<2048, 256, 0, stream>>>((float4*)states,
                                        (long)B_ * S_ * L_ * D_ / 4);
  bucketize<<<B_, 256, 0, stream>>>(group_ids, lengths, pos_emb, seq_id_emb,
                                    empty_tokens, dest, states, mask_out);
  event_mlp_scatter<<<(B_ * H_) / MTILE, 256, 0, stream>>>(
      embed_table, time_gap_emb, seq_id_emb, pos_emb, ln_gamma, ln_beta,
      w1b, b1, w2b, b2, history_tokens, post_tokens, author_tokens,
      action_tokens, time_gap, lengths, dest, states);
}
